// PointerGenerator_73023033966631
// MI455X (gfx1250) — matrix operand order repaired and checked by reading, not compiled, not run on hardware
//
#include <hip/hip_runtime.h>
#include <hip/hip_bf16.h>

// ---------------- problem constants ----------------
#define BB 8
#define TT 128
#define SS 256
#define DD 512
#define VV 32000

typedef _Float16 v16h __attribute__((ext_vector_type(16)));
typedef _Float16 v8h  __attribute__((ext_vector_type(8)));
typedef float    v8f  __attribute__((ext_vector_type(8)));

union AB16 { v16h v; v8h h[2]; };

__device__ __forceinline__ v8f zero8() { v8f c; for (int i = 0; i < 8; ++i) c[i] = 0.f; return c; }

__device__ __forceinline__ v8f wmma32(v16h a, v16h b, v8f c) {
  // v_wmma_f32_16x16x32_f16  (neg_a, A, neg_b, B, c_mod, C, reuse_a, reuse_b)
  return __builtin_amdgcn_wmma_f32_16x16x32_f16(false, a, false, b, (short)0, c, false, false);
}

// A fragment 16x32 f16 from row-major [rows, ld] at (row0..row0+15, k0..k0+31).
// ISA layout: lanes 0-15 row M=lane, halves[0..7]=K 0..7, halves[8..15]=K 16..23;
//             lanes 16-31 same rows, K 8..15 / 24..31.
__device__ __forceinline__ v16h frag_a(const _Float16* p0, int ld, int lane) {
  int m = lane & 15, hi = lane >> 4;
  const _Float16* p = p0 + m * ld + hi * 8;
  AB16 f;
  f.h[0] = *(const v8h*)(p);
  f.h[1] = *(const v8h*)(p + 16);
  return f.v;
}

// B fragment 32x16 f16 where memory holds B transposed: row n, contiguous in K.
// ISA layout: lanes 0-15 col N=lane, halves[0..7]=K 0..7, halves[8..15]=K 16..23; lanes 16-31 K 8..15 / 24..31.
__device__ __forceinline__ v16h frag_b(const _Float16* p0, int ld, int lane) {
  int n = lane & 15, hi = lane >> 4;
  const _Float16* p = p0 + n * ld + hi * 8;
  AB16 f;
  f.h[0] = *(const v8h*)(p);
  f.h[1] = *(const v8h*)(p + 16);
  return f.v;
}

// ---------------- f32 -> f16 convert ----------------
__global__ void k_f32_to_f16(const float* __restrict__ s, _Float16* __restrict__ d, int n) {
  int i = blockIdx.x * blockDim.x + threadIdx.x;
  int stride = gridDim.x * blockDim.x;
  for (; i < n; i += stride) d[i] = (_Float16)s[i];
}

// ---------------- q/k projection: C = A @ W^T + bias, f16 out ----------------
// grid (N/128, M/16), block 256 (8 waves, one 16x16 tile per wave)
__global__ void k_gemm_bias_f16(const _Float16* __restrict__ A, const _Float16* __restrict__ W,
                                const float* __restrict__ bias, _Float16* __restrict__ out) {
  const int K = DD, N = DD;
  int lane = threadIdx.x & 31, w = threadIdx.x >> 5;
  int n0 = blockIdx.x * 128 + w * 16;
  int row0 = blockIdx.y * 16;
  int n = lane & 15, rb = (lane >> 4) * 8;
  float bv = bias[n0 + n];
  v8f c; for (int i = 0; i < 8; ++i) c[i] = bv;
  for (int k0 = 0; k0 < K; k0 += 32) {
    v16h a = frag_a(A + (size_t)row0 * K + k0, K, lane);
    v16h b = frag_b(W + (size_t)n0 * K + k0, K, lane);
    c = wmma32(a, b, c);
  }
  for (int r = 0; r < 8; ++r)
    out[(size_t)(row0 + rb + r) * N + n0 + n] = (_Float16)c[r];
}

// ---------------- flagship: scores = logits_h @ vocab_h^T  [1024 x 32000] ----------------
// grid (V/64, M/64), block 128 (4 waves in 2x2, each owning a 32x32 sub-tile)
__global__ void k_vocab_gemm(const _Float16* __restrict__ Lh, const _Float16* __restrict__ Vh,
                             float* __restrict__ scores) {
  const int K = DD;
  __shared__ _Float16 As[64 * 32];
  __shared__ _Float16 Bs[64 * 32];
  int tid = threadIdx.x;
  int lane = tid & 31, w = tid >> 5;
  int wm = (w >> 1) * 32, wn = (w & 1) * 32;
  int m0 = blockIdx.y * 64, n0 = blockIdx.x * 64;
  int row = tid >> 1, half = tid & 1;
  const _Float16* gA = Lh + (size_t)(m0 + row) * K + half * 16;
  const _Float16* gB = Vh + (size_t)(n0 + row) * K + half * 16;
  v8f c00 = zero8(), c01 = zero8(), c10 = zero8(), c11 = zero8();
  for (int k0 = 0; k0 < K; k0 += 32) {
    *(v8h*)&As[row * 32 + half * 16]     = *(const v8h*)(gA + k0);
    *(v8h*)&As[row * 32 + half * 16 + 8] = *(const v8h*)(gA + k0 + 8);
    *(v8h*)&Bs[row * 32 + half * 16]     = *(const v8h*)(gB + k0);
    *(v8h*)&Bs[row * 32 + half * 16 + 8] = *(const v8h*)(gB + k0 + 8);
    __syncthreads();
    if (k0 + 32 < K) {  // pull next chunk toward L2/L0 (global_prefetch)
      __builtin_prefetch(gA + k0 + 32, 0, 1);
      __builtin_prefetch(gB + k0 + 32, 0, 1);
    }
    v16h a0 = frag_a(&As[(wm)*32],        32, lane);
    v16h a1 = frag_a(&As[(wm + 16) * 32], 32, lane);
    v16h b0 = frag_b(&Bs[(wn)*32],        32, lane);
    v16h b1 = frag_b(&Bs[(wn + 16) * 32], 32, lane);
    c00 = wmma32(a0, b0, c00);
    c01 = wmma32(a0, b1, c01);
    c10 = wmma32(a1, b0, c10);
    c11 = wmma32(a1, b1, c11);
    __syncthreads();
  }
  int n = lane & 15, rb = (lane >> 4) * 8;
  for (int r = 0; r < 8; ++r) {
    int mA = m0 + wm + rb + r, mB = mA + 16;
    scores[(size_t)mA * VV + n0 + wn + n]      = c00[r];
    scores[(size_t)mA * VV + n0 + wn + 16 + n] = c01[r];
    scores[(size_t)mB * VV + n0 + wn + n]      = c10[r];
    scores[(size_t)mB * VV + n0 + wn + 16 + n] = c11[r];
  }
}

// ---------------- fused pointer path (f32): scores_S, softmax_S, text_vec, switch --------
// grid (T/16, B), block 256
__global__ void k_pointer(const _Float16* __restrict__ Qh, const _Float16* __restrict__ Kh,
                          const float* __restrict__ E, const float* __restrict__ logits,
                          const float* __restrict__ tgt, const unsigned char* __restrict__ mask,
                          const float* __restrict__ Wp, const float* __restrict__ bp,
                          float* __restrict__ attnBuf, float* __restrict__ swBuf,
                          float* __restrict__ lswBuf) {
  __shared__ float qs[16 * DD];
  __shared__ float sc[16 * SS];
  __shared__ float red[256];
  int b = blockIdx.y, t0 = blockIdx.x * 16;
  int tid = threadIdx.x;
  for (int i = tid; i < 16 * DD; i += 256)
    qs[i] = (float)Qh[(size_t)(b * TT + t0 + (i >> 9)) * DD + (i & (DD - 1))];
  __syncthreads();
  const float rsD = 0.04419417382415922f;  // 1/sqrt(512)
  for (int t = 0; t < 16; ++t) {
    const _Float16* krow = Kh + (size_t)(b * SS + tid) * DD;
    float acc = 0.f;
    for (int d = 0; d < DD; ++d) acc += qs[t * DD + d] * (float)krow[d];
    sc[t * SS + tid] = mask[b * SS + tid] ? acc * rsD : -1e9f;
  }
  __syncthreads();
  for (int t = 0; t < 16; ++t) {
    float v = sc[t * SS + tid];
    red[tid] = v; __syncthreads();
    for (int o = 128; o > 0; o >>= 1) { if (tid < o) red[tid] = fmaxf(red[tid], red[tid + o]); __syncthreads(); }
    float mx = red[0]; __syncthreads();
    float e = expf(v - mx);
    red[tid] = e; __syncthreads();
    for (int o = 128; o > 0; o >>= 1) { if (tid < o) red[tid] += red[tid + o]; __syncthreads(); }
    float z = red[0]; __syncthreads();
    float a = e / z;
    sc[t * SS + tid] = a;
    attnBuf[(size_t)(b * TT + t0 + t) * SS + tid] = a;
  }
  __syncthreads();
  for (int t = 0; t < 16; ++t) {
    int m = b * TT + t0 + t;
    float part = 0.f;
    for (int rep = 0; rep < 2; ++rep) {
      int d = tid + rep * 256;
      float tv = 0.f;
      for (int s = 0; s < SS; ++s) tv += sc[t * SS + s] * E[((size_t)b * SS + s) * DD + d];
      part += Wp[DD + d] * tv + Wp[d] * logits[(size_t)m * DD + d] + Wp[2 * DD + d] * tgt[(size_t)m * DD + d];
    }
    red[tid] = part; __syncthreads();
    for (int o = 128; o > 0; o >>= 1) { if (tid < o) red[tid] += red[tid + o]; __syncthreads(); }
    if (tid == 0) {
      float sw = 1.f / (1.f + expf(-(red[0] + bp[0])));
      swBuf[m] = sw;
      lswBuf[m] = logf(sw);
    }
    __syncthreads();
  }
}

// ---------------- per-row max / sum-exp over V (L2-resident) ----------------
__global__ void k_rowstats(const float* __restrict__ scores, float* __restrict__ rowM,
                           float* __restrict__ rowZ, float* __restrict__ rowLZ) {
  __shared__ float red[256];
  int m = blockIdx.x, tid = threadIdx.x;
  const float* row = scores + (size_t)m * VV;
  float mx = -3.4e38f;
  for (int v = tid; v < VV; v += 256) mx = fmaxf(mx, row[v]);
  red[tid] = mx; __syncthreads();
  for (int o = 128; o > 0; o >>= 1) { if (tid < o) red[tid] = fmaxf(red[tid], red[tid + o]); __syncthreads(); }
  mx = red[0]; __syncthreads();
  float s = 0.f;
  for (int v = tid; v < VV; v += 256) s += expf(row[v] - mx);
  red[tid] = s; __syncthreads();
  for (int o = 128; o > 0; o >>= 1) { if (tid < o) red[tid] += red[tid + o]; __syncthreads(); }
  if (tid == 0) { rowM[m] = mx; rowZ[m] = red[0]; rowLZ[m] = logf(red[0]); }
}

// ---------------- bulk combine: out = log(sw) + score - M - logZ ----------------
// grid (V/256, 1024)
__global__ void k_combine(const float* __restrict__ scores, const float* __restrict__ rowM,
                          const float* __restrict__ rowLZ, const float* __restrict__ lsw,
                          float* __restrict__ out) {
  int m = blockIdx.y;
  int v = blockIdx.x * 256 + threadIdx.x;
  size_t i = (size_t)m * VV + v;
  out[i] = lsw[m] + scores[i] - rowM[m] - rowLZ[m];
}

// ---------------- scatter correction for the <=256 touched vocab ids per (b,t) ----------
// grid (T, B), block 256
__global__ void k_scatter(const float* __restrict__ scores, const float* __restrict__ attnBuf,
                          const int* __restrict__ text, const float* __restrict__ rowM,
                          const float* __restrict__ rowZ, const float* __restrict__ swBuf,
                          float* __restrict__ out) {
  __shared__ int tx[SS];
  __shared__ float at[SS];
  int b = blockIdx.y, t = blockIdx.x, s = threadIdx.x;
  int m = b * TT + t;
  tx[s] = text[b * SS + s];
  at[s] = attnBuf[(size_t)m * SS + s];
  __syncthreads();
  int v = tx[s];
  bool owner = true;
  for (int j = 0; j < s; ++j)
    if (tx[j] == v) { owner = false; break; }
  if (owner) {
    float val = 0.f;
    for (int j = 0; j < SS; ++j)
      if (tx[j] == v) val += at[j];
    float sw = swBuf[m];
    float pv = expf(scores[(size_t)m * VV + v] - rowM[m]) / rowZ[m];
    out[(size_t)m * VV + v] = logf((1.f - sw) * val + sw * pv);
  }
}

// ---------------- host ----------------
extern "C" void kernel_launch(void* const* d_in, const int* in_sizes, int n_in,
                              void* d_out, int out_size, void* d_ws, size_t ws_size,
                              hipStream_t stream) {
  (void)in_sizes; (void)n_in; (void)out_size; (void)ws_size;
  const float* logits = (const float*)d_in[0];
  const float* etext  = (const float*)d_in[1];
  const float* etgt   = (const float*)d_in[2];
  const int*   text   = (const int*)d_in[3];
  const unsigned char* mask = (const unsigned char*)d_in[4];
  const float* vg = (const float*)d_in[5];
  const float* Wq = (const float*)d_in[6];
  const float* bq = (const float*)d_in[7];
  const float* Wk = (const float*)d_in[8];
  const float* bk = (const float*)d_in[9];
  const float* Wp = (const float*)d_in[10];
  const float* bp = (const float*)d_in[11];
  float* out = (float*)d_out;

  char* w = (char*)d_ws;
  size_t off = 0;
  auto carve = [&](size_t bytes) { void* p = w + off; off += (bytes + 255) & ~(size_t)255; return p; };
  float*     scoresV = (float*)carve((size_t)BB * TT * VV * 4);  // 131 MB
  float*     attnBuf = (float*)carve((size_t)BB * TT * SS * 4);
  _Float16*  Qh  = (_Float16*)carve((size_t)BB * TT * DD * 2);
  _Float16*  Kh  = (_Float16*)carve((size_t)BB * SS * DD * 2);
  _Float16*  Lh  = (_Float16*)carve((size_t)BB * TT * DD * 2);
  _Float16*  Eh  = (_Float16*)carve((size_t)BB * SS * DD * 2);
  _Float16*  Vh  = (_Float16*)carve((size_t)VV * DD * 2);
  _Float16*  Wqh = (_Float16*)carve((size_t)DD * DD * 2);
  _Float16*  Wkh = (_Float16*)carve((size_t)DD * DD * 2);
  float*     rowM  = (float*)carve(BB * TT * 4);
  float*     rowZ  = (float*)carve(BB * TT * 4);
  float*     rowLZ = (float*)carve(BB * TT * 4);
  float*     swBuf = (float*)carve(BB * TT * 4);
  float*     lswBuf = (float*)carve(BB * TT * 4);

  // 1) f16 operand copies
  k_f32_to_f16<<<512, 256, 0, stream>>>(logits, Lh, BB * TT * DD);
  k_f32_to_f16<<<512, 256, 0, stream>>>(etext, Eh, BB * SS * DD);
  k_f32_to_f16<<<2048, 256, 0, stream>>>(vg, Vh, VV * DD);
  k_f32_to_f16<<<256, 256, 0, stream>>>(Wq, Wqh, DD * DD);
  k_f32_to_f16<<<256, 256, 0, stream>>>(Wk, Wkh, DD * DD);

  // 2) q / k projections (WMMA)
  k_gemm_bias_f16<<<dim3(DD / 128, BB * TT / 16), 256, 0, stream>>>(Lh, Wqh, bq, Qh);
  k_gemm_bias_f16<<<dim3(DD / 128, BB * SS / 16), 256, 0, stream>>>(Eh, Wkh, bk, Kh);

  // 3) vocab GEMM (WMMA flagship)
  k_vocab_gemm<<<dim3(VV / 64, BB * TT / 64), 128, 0, stream>>>(Lh, Vh, scoresV);

  // 4) pointer path (f32 fused)
  k_pointer<<<dim3(TT / 16, BB), 256, 0, stream>>>(Qh, Kh, etext, logits, etgt, mask, Wp, bp,
                                                   attnBuf, swBuf, lswBuf);

  // 5) softmax stats over V
  k_rowstats<<<BB * TT, 256, 0, stream>>>(scoresV, rowM, rowZ, rowLZ);

  // 6) bulk log-combine, then scatter correction for touched vocab ids
  k_combine<<<dim3(VV / 256, BB * TT), 256, 0, stream>>>(scoresV, rowM, rowLZ, lswBuf, out);
  k_scatter<<<dim3(TT, BB), 256, 0, stream>>>(scoresV, attnBuf, text, rowM, rowZ, swBuf, out);

  (void)text; (void)mask;
}
